// ImageCaptionModel_81200651698683
// MI455X (gfx1250) — compile-verified, matrix operand order checked by reading, not run on hardware
//
#include <hip/hip_runtime.h>
#include <hip/hip_bf16.h>

// ---------------------------------------------------------------------------
// Types for CDNA5 WMMA bf16 path
// ---------------------------------------------------------------------------
typedef __bf16 v16bf __attribute__((ext_vector_type(16)));
typedef __bf16 v8bf  __attribute__((ext_vector_type(8)));
typedef float  v8f   __attribute__((ext_vector_type(8)));

#define T_STEPS 40
#define BATCH   128
#define HID     512
#define EMB     512
#define CFEAT   2048
#define VOCAB   10000

__device__ __forceinline__ unsigned short f2bf(float f) {
    unsigned int u = __float_as_uint(f);
    unsigned int r = u + 0x7fffu + ((u >> 16) & 1u);
    return (unsigned short)(r >> 16);
}
__device__ __forceinline__ float bf2f(unsigned short h) {
    return __uint_as_float(((unsigned int)h) << 16);
}
__device__ __forceinline__ float sigf(float x) {
    return 1.0f / (1.0f + __expf(-x));
}

// Load one 16x32 bf16 WMMA fragment for this lane.
// Layout (ISA 7.12.2, 16-bit A 16x32): lanes 0-15 hold row (l16), K chunks
// {k0..k0+7, k0+16..k0+23} with k0 = kk + 8*half.  Matrix is row-major
// [rows x ld] of bf16 (stored as ushort).  B fragments use the mirrored
// layout on an [N x K] (transposed) weight matrix.
__device__ __forceinline__ v16bf load_frag(const unsigned short* __restrict__ base,
                                           int ld, int row, int k0) {
    union U { v16bf v; struct S { v8bf lo, hi; } s; } u;
    const unsigned short* p = base + (size_t)row * ld + k0;
    u.s.lo = *reinterpret_cast<const v8bf*>(p);
    u.s.hi = *reinterpret_cast<const v8bf*>(p + 16);
    return u.v;
}

__device__ __forceinline__ v8f wmma_bf16(v16bf a, v16bf b, v8f c) {
    return __builtin_amdgcn_wmma_f32_16x16x32_bf16(false, a, false, b,
                                                   (short)0, c, false, false);
}

// ---------------------------------------------------------------------------
// Elementwise helpers
// ---------------------------------------------------------------------------
__global__ void f32_to_bf16_kernel(const float* __restrict__ in,
                                   unsigned short* __restrict__ out, int n) {
    int i = blockIdx.x * blockDim.x + threadIdx.x;
    int stride = gridDim.x * blockDim.x;
    for (; i < n; i += stride) out[i] = f2bf(in[i]);
}

// Wt[n*K + k] = bf16(W[k*N + n])   (W is [K,N] row-major fp32)
__global__ void transpose_bf16_kernel(const float* __restrict__ W,
                                      unsigned short* __restrict__ Wt,
                                      int K, int N) {
    int i = blockIdx.x * blockDim.x + threadIdx.x;
    int stride = gridDim.x * blockDim.x;
    int total = K * N;
    for (; i < total; i += stride) {
        int n = i / K;
        int k = i - n * K;
        Wt[i] = f2bf(W[(size_t)k * N + n]);
    }
}

__global__ void zero_bf16_kernel(unsigned short* __restrict__ p, int n) {
    int i = blockIdx.x * blockDim.x + threadIdx.x;
    int stride = gridDim.x * blockDim.x;
    for (; i < n; i += stride) p[i] = 0;
}

// X0[t][b][0:512]   = bf16(emb[tok[b][t]])
// X0[t][b][512:1024]= proc_bf16[b]
__global__ void build_x0_kernel(const float* __restrict__ emb,
                                const int* __restrict__ tok,
                                const unsigned short* __restrict__ proc,
                                unsigned short* __restrict__ X0) {
    int i = blockIdx.x * blockDim.x + threadIdx.x;
    int stride = gridDim.x * blockDim.x;
    const int total = T_STEPS * BATCH * (EMB + HID);
    for (; i < total; i += stride) {
        int t = i / (BATCH * (EMB + HID));
        int r = i - t * (BATCH * (EMB + HID));
        int b = r / (EMB + HID);
        int c = r - b * (EMB + HID);
        if (c < EMB) {
            int v = tok[b * T_STEPS + t];
            X0[i] = f2bf(emb[(size_t)v * EMB + c]);
        } else {
            X0[i] = proc[b * HID + (c - EMB)];
        }
    }
}

__global__ void hfinal_kernel(const unsigned short* __restrict__ h0,
                              const unsigned short* __restrict__ h1,
                              float* __restrict__ out) {
    int i = blockIdx.x * blockDim.x + threadIdx.x;
    const int n = BATCH * HID;
    if (i < n) {
        out[i]     = bf2f(h0[i]);
        out[n + i] = bf2f(h1[i]);
    }
}

// ---------------------------------------------------------------------------
// Generic bf16 WMMA GEMM:  out = act(A[MxK] @ Bt[NxK]^T + bias)
// Each wave computes an (MT*16) x (NT*16) super-tile: A fragments are reused
// across NT WMMAs and B fragments across MT WMMAs (register-level reuse).
// Super-tile id is wave-uniform, so EXEC stays all-ones across the WMMAs.
// remapB > 0: output row m = t*remapB + b is stored at row b*remapT + t.
// fp32 output stores are non-temporal: the logits stream (205 MB) must not
// evict the L2-resident bf16 weights on its way to HBM.
// ---------------------------------------------------------------------------
template<int MT, int NT>
__global__ __launch_bounds__(256) void gemm_bf16_kernel(
    const unsigned short* __restrict__ A, int lda,
    const unsigned short* __restrict__ Bt, int ldb,
    const float* __restrict__ bias,
    float* __restrict__ outF, unsigned short* __restrict__ outB, int ldo,
    int M, int N, int K, int act, int remapB, int remapT) {
    int lane = threadIdx.x & 31;
    int half = lane >> 4;
    int l16  = lane & 15;
    int nSupN = (N >> 4) / NT;
    int nSup  = ((M >> 4) / MT) * nSupN;
    int wid    = (blockIdx.x * blockDim.x + threadIdx.x) >> 5;
    int nWaves = (gridDim.x * blockDim.x) >> 5;
    for (int tile = wid; tile < nSup; tile += nWaves) {
        int m0 = (tile / nSupN) * (MT * 16);
        int n0 = (tile % nSupN) * (NT * 16);
        v8f c[MT][NT];
        #pragma unroll
        for (int i = 0; i < MT; ++i)
            #pragma unroll
            for (int j = 0; j < NT; ++j)
                c[i][j] = v8f{0.f, 0.f, 0.f, 0.f, 0.f, 0.f, 0.f, 0.f};
        for (int kk = 0; kk < K; kk += 32) {
            int k0 = kk + (half << 3);
            v16bf a[MT];
            #pragma unroll
            for (int i = 0; i < MT; ++i)
                a[i] = load_frag(A, lda, m0 + i * 16 + l16, k0);
            #pragma unroll
            for (int j = 0; j < NT; ++j) {
                v16bf b = load_frag(Bt, ldb, n0 + j * 16 + l16, k0);
                #pragma unroll
                for (int i = 0; i < MT; ++i)
                    c[i][j] = wmma_bf16(a[i], b, c[i][j]);
            }
        }
        #pragma unroll
        for (int j = 0; j < NT; ++j) {
            int n = n0 + j * 16 + l16;
            float bv = bias ? bias[n] : 0.f;
            #pragma unroll
            for (int i = 0; i < MT; ++i) {
                #pragma unroll
                for (int r = 0; r < 8; ++r) {
                    int m = m0 + i * 16 + r + (half << 3);
                    float v = c[i][j][r] + bv;
                    if (act == 1) v = (v >= 0.f) ? v : 0.01f * v;   // LeakyReLU
                    size_t orow = (size_t)m;
                    if (remapB > 0) {
                        int t  = m / remapB;
                        int bb = m - t * remapB;
                        orow = (size_t)bb * remapT + t;
                    }
                    if (outB) outB[orow * ldo + n] = f2bf(v);
                    else      __builtin_nontemporal_store(v, &outF[orow * ldo + n]);
                }
            }
        }
    }
}

// ---------------------------------------------------------------------------
// GRU z/r kernel: xh = [X | H] (concat along K),
//   z = sigmoid(xh @ Wu + bu),  rh = sigmoid(xh @ Wr + br) * h
// 32x32 super-tile per wave; each A fragment feeds 4 WMMAs (Wu and Wr for
// 2 N sub-tiles), each weight fragment feeds 2 WMMAs.
// The concat K loop is split into two straight-line ranges (X then H) so the
// hot loop has no per-iteration address select/branch.
// M=128, N=512 fixed -> 4 x 16 = 64 super-tiles.
// ---------------------------------------------------------------------------
__global__ __launch_bounds__(256) void gru_zr_kernel(
    const unsigned short* __restrict__ X, int Kx,
    const unsigned short* __restrict__ H,
    const unsigned short* __restrict__ Wut,
    const unsigned short* __restrict__ Wrt, int ldb,
    const float* __restrict__ bu, const float* __restrict__ br,
    float* __restrict__ Z, unsigned short* __restrict__ RH) {
    constexpr int MT = 2, NT = 2;
    int lane = threadIdx.x & 31;
    int half = lane >> 4;
    int l16  = lane & 15;
    int wid    = (blockIdx.x * blockDim.x + threadIdx.x) >> 5;
    int nWaves = (gridDim.x * blockDim.x) >> 5;
    const int nSupN = (HID >> 4) / NT;                     // 16
    const int nSup  = ((BATCH >> 4) / MT) * nSupN;         // 64
    for (int tile = wid; tile < nSup; tile += nWaves) {
        int m0 = (tile / nSupN) * (MT * 16);
        int n0 = (tile % nSupN) * (NT * 16);
        v8f cu[MT][NT], cr[MT][NT];
        #pragma unroll
        for (int i = 0; i < MT; ++i)
            #pragma unroll
            for (int j = 0; j < NT; ++j) {
                cu[i][j] = v8f{0.f, 0.f, 0.f, 0.f, 0.f, 0.f, 0.f, 0.f};
                cr[i][j] = v8f{0.f, 0.f, 0.f, 0.f, 0.f, 0.f, 0.f, 0.f};
            }
        // ---- K range 1: A fragments from X, weight cols [0, Kx) ----
        for (int kk = 0; kk < Kx; kk += 32) {
            int k0 = kk + (half << 3);
            v16bf a[MT];
            #pragma unroll
            for (int i = 0; i < MT; ++i)
                a[i] = load_frag(X, Kx, m0 + i * 16 + l16, k0);
            #pragma unroll
            for (int j = 0; j < NT; ++j) {
                int brow = n0 + j * 16 + l16;
                v16bf wu = load_frag(Wut, ldb, brow, k0);
                v16bf wr = load_frag(Wrt, ldb, brow, k0);
                #pragma unroll
                for (int i = 0; i < MT; ++i) {
                    cu[i][j] = wmma_bf16(a[i], wu, cu[i][j]);
                    cr[i][j] = wmma_bf16(a[i], wr, cr[i][j]);
                }
            }
        }
        // ---- K range 2: A fragments from H, weight cols [Kx, Kx+HID) ----
        for (int kk = 0; kk < HID; kk += 32) {
            int k0 = kk + (half << 3);
            v16bf a[MT];
            #pragma unroll
            for (int i = 0; i < MT; ++i)
                a[i] = load_frag(H, HID, m0 + i * 16 + l16, k0);
            #pragma unroll
            for (int j = 0; j < NT; ++j) {
                int brow = n0 + j * 16 + l16;
                v16bf wu = load_frag(Wut, ldb, brow, Kx + k0);
                v16bf wr = load_frag(Wrt, ldb, brow, Kx + k0);
                #pragma unroll
                for (int i = 0; i < MT; ++i) {
                    cu[i][j] = wmma_bf16(a[i], wu, cu[i][j]);
                    cr[i][j] = wmma_bf16(a[i], wr, cr[i][j]);
                }
            }
        }
        #pragma unroll
        for (int j = 0; j < NT; ++j) {
            int n = n0 + j * 16 + l16;
            float buv = bu[n], brv = br[n];
            #pragma unroll
            for (int i = 0; i < MT; ++i) {
                #pragma unroll
                for (int r = 0; r < 8; ++r) {
                    int m   = m0 + i * 16 + r + (half << 3);
                    int idx = m * HID + n;
                    float z  = sigf(cu[i][j][r] + buv);
                    float rr = sigf(cr[i][j][r] + brv);
                    Z[idx]  = z;
                    RH[idx] = f2bf(rr * bf2f(H[idx]));
                }
            }
        }
    }
}

// ---------------------------------------------------------------------------
// GRU candidate + state update:
//   hh = tanh([X | RH] @ W + b);  h = z*h + (1-z)*hh   (in place, bf16)
// 32x64 super-tile per wave (A fragment reused by 4 N sub-tiles), split
// concat K loop.  Optionally appends new h to history for the logits GEMM.
// ---------------------------------------------------------------------------
__global__ __launch_bounds__(256) void gru_hhat_kernel(
    const unsigned short* __restrict__ X, int Kx,
    const unsigned short* __restrict__ RH,
    const unsigned short* __restrict__ Wt, int ldb,
    const float* __restrict__ b,
    const float* __restrict__ Z,
    unsigned short* __restrict__ Hst,
    unsigned short* __restrict__ hist) {
    constexpr int MT = 2, NT = 4;
    int lane = threadIdx.x & 31;
    int half = lane >> 4;
    int l16  = lane & 15;
    int wid    = (blockIdx.x * blockDim.x + threadIdx.x) >> 5;
    int nWaves = (gridDim.x * blockDim.x) >> 5;
    const int nSupN = (HID >> 4) / NT;                     // 8
    const int nSup  = ((BATCH >> 4) / MT) * nSupN;         // 32
    for (int tile = wid; tile < nSup; tile += nWaves) {
        int m0 = (tile / nSupN) * (MT * 16);
        int n0 = (tile % nSupN) * (NT * 16);
        v8f c[MT][NT];
        #pragma unroll
        for (int i = 0; i < MT; ++i)
            #pragma unroll
            for (int j = 0; j < NT; ++j)
                c[i][j] = v8f{0.f, 0.f, 0.f, 0.f, 0.f, 0.f, 0.f, 0.f};
        // ---- K range 1: A from X ----
        for (int kk = 0; kk < Kx; kk += 32) {
            int k0 = kk + (half << 3);
            v16bf a[MT];
            #pragma unroll
            for (int i = 0; i < MT; ++i)
                a[i] = load_frag(X, Kx, m0 + i * 16 + l16, k0);
            #pragma unroll
            for (int j = 0; j < NT; ++j) {
                v16bf w = load_frag(Wt, ldb, n0 + j * 16 + l16, k0);
                #pragma unroll
                for (int i = 0; i < MT; ++i)
                    c[i][j] = wmma_bf16(a[i], w, c[i][j]);
            }
        }
        // ---- K range 2: A from RH ----
        for (int kk = 0; kk < HID; kk += 32) {
            int k0 = kk + (half << 3);
            v16bf a[MT];
            #pragma unroll
            for (int i = 0; i < MT; ++i)
                a[i] = load_frag(RH, HID, m0 + i * 16 + l16, k0);
            #pragma unroll
            for (int j = 0; j < NT; ++j) {
                v16bf w = load_frag(Wt, ldb, n0 + j * 16 + l16, Kx + k0);
                #pragma unroll
                for (int i = 0; i < MT; ++i)
                    c[i][j] = wmma_bf16(a[i], w, c[i][j]);
            }
        }
        #pragma unroll
        for (int j = 0; j < NT; ++j) {
            int n = n0 + j * 16 + l16;
            float bv = b[n];
            #pragma unroll
            for (int i = 0; i < MT; ++i) {
                #pragma unroll
                for (int r = 0; r < 8; ++r) {
                    int m   = m0 + i * 16 + r + (half << 3);
                    int idx = m * HID + n;
                    float hh = tanhf(c[i][j][r] + bv);
                    float z  = Z[idx];
                    float hn = z * bf2f(Hst[idx]) + (1.f - z) * hh;
                    unsigned short hb = f2bf(hn);
                    Hst[idx] = hb;
                    if (hist) hist[idx] = hb;
                }
            }
        }
    }
}

// ---------------------------------------------------------------------------
// Host orchestration
// ---------------------------------------------------------------------------
extern "C" void kernel_launch(void* const* d_in, const int* in_sizes, int n_in,
                              void* d_out, int out_size, void* d_ws, size_t ws_size,
                              hipStream_t stream) {
    (void)in_sizes; (void)n_in; (void)out_size; (void)ws_size;
    const float* cnn   = (const float*)d_in[0];
    const int*   tok   = (const int*)  d_in[1];
    const float* emb   = (const float*)d_in[3];
    const float* inW   = (const float*)d_in[4];
    const float* inB   = (const float*)d_in[5];
    const float* outW  = (const float*)d_in[6];
    const float* outB  = (const float*)d_in[7];
    const float* Wu0   = (const float*)d_in[8];   const float* bu0 = (const float*)d_in[9];
    const float* Wr0   = (const float*)d_in[10];  const float* br0 = (const float*)d_in[11];
    const float* W0    = (const float*)d_in[12];  const float* b0  = (const float*)d_in[13];
    const float* Wu1   = (const float*)d_in[14];  const float* bu1 = (const float*)d_in[15];
    const float* Wr1   = (const float*)d_in[16];  const float* br1 = (const float*)d_in[17];
    const float* W1    = (const float*)d_in[18];  const float* b1  = (const float*)d_in[19];

    float* dout = (float*)d_out;

    // ---- workspace carve-up (256-byte aligned) ----
    unsigned char* ws = (unsigned char*)d_ws;
    size_t off = 0;
    auto carve = [&](size_t bytes) -> unsigned char* {
        unsigned char* p = ws + off;
        off += (bytes + 255) & ~(size_t)255;
        return p;
    };
    const int HP0 = EMB + HID + HID;   // 1536
    const int HP1 = HID + HID;         // 1024
    unsigned short* inWt  = (unsigned short*)carve((size_t)HID * CFEAT * 2);
    unsigned short* Wu0t  = (unsigned short*)carve((size_t)HID * HP0 * 2);
    unsigned short* Wr0t  = (unsigned short*)carve((size_t)HID * HP0 * 2);
    unsigned short* W0t   = (unsigned short*)carve((size_t)HID * HP0 * 2);
    unsigned short* Wu1t  = (unsigned short*)carve((size_t)HID * HP1 * 2);
    unsigned short* Wr1t  = (unsigned short*)carve((size_t)HID * HP1 * 2);
    unsigned short* W1t   = (unsigned short*)carve((size_t)HID * HP1 * 2);
    unsigned short* outWt = (unsigned short*)carve((size_t)VOCAB * HID * 2);
    unsigned short* cnnb  = (unsigned short*)carve((size_t)BATCH * CFEAT * 2);
    unsigned short* procb = (unsigned short*)carve((size_t)BATCH * HID * 2);
    unsigned short* X0    = (unsigned short*)carve((size_t)T_STEPS * BATCH * (EMB + HID) * 2);
    unsigned short* h0    = (unsigned short*)carve((size_t)BATCH * HID * 2);
    unsigned short* h1    = (unsigned short*)carve((size_t)BATCH * HID * 2);
    float*          Zb    = (float*)         carve((size_t)BATCH * HID * 4);
    unsigned short* RHb   = (unsigned short*)carve((size_t)BATCH * HID * 2);
    unsigned short* H1    = (unsigned short*)carve((size_t)T_STEPS * BATCH * HID * 2);

    // ---- 1. weight/activation conversion to bf16 (transposed [N,K]) ----
    {
        int n = BATCH * CFEAT;
        f32_to_bf16_kernel<<<(n + 255) / 256, 256, 0, stream>>>(cnn, cnnb, n);
    }
    transpose_bf16_kernel<<<4096, 256, 0, stream>>>(inW,  inWt,  CFEAT, HID);
    transpose_bf16_kernel<<<3072, 256, 0, stream>>>(Wu0,  Wu0t,  HP0,   HID);
    transpose_bf16_kernel<<<3072, 256, 0, stream>>>(Wr0,  Wr0t,  HP0,   HID);
    transpose_bf16_kernel<<<3072, 256, 0, stream>>>(W0,   W0t,   HP0,   HID);
    transpose_bf16_kernel<<<2048, 256, 0, stream>>>(Wu1,  Wu1t,  HP1,   HID);
    transpose_bf16_kernel<<<2048, 256, 0, stream>>>(Wr1,  Wr1t,  HP1,   HID);
    transpose_bf16_kernel<<<2048, 256, 0, stream>>>(W1,   W1t,   HP1,   HID);
    transpose_bf16_kernel<<<8192, 256, 0, stream>>>(outW, outWt, HID,   VOCAB);

    // ---- 2. proc = leakyrelu(cnn @ input_W + input_b)  [128,512] bf16 ----
    gemm_bf16_kernel<2, 2><<<8, 256, 0, stream>>>(cnnb, CFEAT, inWt, CFEAT, inB,
                                                  nullptr, procb, HID,
                                                  BATCH, HID, CFEAT, /*act=*/1, 0, 0);

    // ---- 3. X0[t] = [emb(tok[:,t]) | proc]  for all t ----
    build_x0_kernel<<<8192, 256, 0, stream>>>(emb, tok, procb, X0);

    // ---- 4. zero initial hidden state ----
    zero_bf16_kernel<<<256, 256, 0, stream>>>(h0, BATCH * HID);
    zero_bf16_kernel<<<256, 256, 0, stream>>>(h1, BATCH * HID);

    // ---- 5. recurrent GRU, 2 cells x 40 steps ----
    for (int t = 0; t < T_STEPS; ++t) {
        const unsigned short* x0t = X0 + (size_t)t * BATCH * (EMB + HID);
        // cell 0: x = [emb_t | proc] (Kx = 1024)
        gru_zr_kernel<<<8, 256, 0, stream>>>(x0t, EMB + HID, h0,
                                             Wu0t, Wr0t, HP0, bu0, br0, Zb, RHb);
        gru_hhat_kernel<<<4, 256, 0, stream>>>(x0t, EMB + HID, RHb,
                                               W0t, HP0, b0, Zb, h0, nullptr);
        // cell 1: x = h0 (Kx = 512)
        gru_zr_kernel<<<8, 256, 0, stream>>>(h0, HID, h1,
                                             Wu1t, Wr1t, HP1, bu1, br1, Zb, RHb);
        gru_hhat_kernel<<<4, 256, 0, stream>>>(h0, HID, RHb,
                                               W1t, HP1, b1, Zb, h1,
                                               H1 + (size_t)t * BATCH * HID);
    }

    // ---- 6. logits = H1[5120,512] @ outWt^T + out_b, rows remapped t,b -> b,t
    //      MT=4: each out_W fragment feeds 4 WMMAs.  50000 super-tiles,
    //      non-temporal fp32 stores straight to HBM.
    gemm_bf16_kernel<4, 1><<<2048, 256, 0, stream>>>(H1, HID, outWt, HID, outB,
                                                     dout, nullptr, VOCAB,
                                                     T_STEPS * BATCH, VOCAB, HID,
                                                     /*act=*/0, /*remapB=*/BATCH,
                                                     /*remapT=*/T_STEPS);

    // ---- 7. h_final [2,128,512] fp32 after the logits block ----
    hfinal_kernel<<<(BATCH * HID + 255) / 256, 256, 0, stream>>>(
        h0, h1, dout + (size_t)BATCH * T_STEPS * VOCAB);
}
